// AgentSelfAttention_47296179863844
// MI455X (gfx1250) — matrix operand order repaired and checked by reading, not compiled, not run on hardware
//
#include <hip/hip_runtime.h>
#include <hip/hip_bf16.h>
#include <stdint.h>
#include <stddef.h>

// ---------------------------------------------------------------------------
// AgentSelfAttention on MI455X (gfx1250): all matmuls via v_wmma_f32_16x16x32_bf16
// __launch_bounds__(256,1) everywhere: 8 waves/WGP -> 2 waves/SIMD -> up to
// 512 VGPRs/wave, so 64x64 wave tiles (128 acc VGPRs) stay in registers.
// ---------------------------------------------------------------------------

typedef __bf16 bf16_t;
typedef __attribute__((ext_vector_type(16))) __bf16 v16bf;
typedef __attribute__((ext_vector_type(8)))  __bf16 v8bf;
typedef __attribute__((ext_vector_type(8)))  float  v8f;

#define NBATCH 4
#define NTOK   4096
#define DIM    1024
#define NHEAD  16
#define DHEAD  64
#define MAGENT 128
#define NQKV   3072
#define MROWS  (NBATCH * NTOK)   // 16384

// D = A*B + C, 16x16x32 bf16 -> f32
#define WMMA_BF16(a, b, c) \
  __builtin_amdgcn_wmma_f32_16x16x32_bf16(false, (a), false, (b), (short)0, (c), false, false)

// ---- fragment loaders ------------------------------------------------------
// A-matrix 16x32 bf16 layout (matches B): lane holds row M=lane&15;
// kb=(lane>>4)*16; elements hold K = kb..kb+15 contiguous.
static __device__ __forceinline__ v16bf ldA_bf16(const bf16_t* p /* row + k0 + kb */) {
  v16bf f;
#pragma unroll
  for (int i = 0; i < 16; ++i) f[i] = p[i];
  return f;
}
static __device__ __forceinline__ v16bf ldA_f32(const float* p /* row + k0 + kb */) {
  v16bf f;
#pragma unroll
  for (int i = 0; i < 16; ++i) f[i] = (bf16_t)p[i];
  return f;
}
// B-matrix 32x16 bf16: lane holds col N=lane&15; K = (lane>>4)*16 .. +15,
// sequential -> one contiguous 32B load from an N-major (transposed) buffer.
// caller passes p = Bt + col*K + k0 + kb2  with kb2=(lane>>4)*16.

// ---- 16-lane half-group row reductions (C-layout softmax stats) ------------
static __device__ __forceinline__ float hmax16(float v) {
  v = fmaxf(v, __shfl_xor(v, 1, 32));
  v = fmaxf(v, __shfl_xor(v, 2, 32));
  v = fmaxf(v, __shfl_xor(v, 4, 32));
  v = fmaxf(v, __shfl_xor(v, 8, 32));
  return v;
}
static __device__ __forceinline__ float hsum16(float v) {
  v += __shfl_xor(v, 1, 32);
  v += __shfl_xor(v, 2, 32);
  v += __shfl_xor(v, 4, 32);
  v += __shfl_xor(v, 8, 32);
  return v;
}

// ---------------------------------------------------------------------------
// K1/K2: LDS-tiled transpose + fp32->bf16 convert.  in: rows x cols (f32),
// out: cols x rows (bf16, N-major for WMMA B-fragments).
// ---------------------------------------------------------------------------
__global__ __launch_bounds__(256, 1)
void k_transpose_bf16(const float* __restrict__ in, bf16_t* __restrict__ out,
                      int rows, int cols) {
  __shared__ bf16_t tile[32][33];
  const int bx = blockIdx.x * 32;  // col base
  const int by = blockIdx.y * 32;  // row base
  const int tx = threadIdx.x, ty = threadIdx.y;  // 32 x 8
#pragma unroll
  for (int i = ty; i < 32; i += 8) {
    int r = by + i, c = bx + tx;
    tile[i][tx] = (r < rows && c < cols) ? (bf16_t)in[(size_t)r * cols + c] : (bf16_t)0.0f;
  }
  __syncthreads();
#pragma unroll
  for (int i = ty; i < 32; i += 8) {
    int r = bx + i, c = by + tx;   // out is [cols][rows]
    if (r < cols && c < rows) out[(size_t)r * rows + c] = tile[tx][i];
  }
}

// K3: agent tokens * SCALE -> bf16
__global__ __launch_bounds__(256, 1)
void k_scale_bf16(const float* __restrict__ in, bf16_t* __restrict__ out,
                  int n, float scale) {
  int i = blockIdx.x * blockDim.x + threadIdx.x;
  if (i < n) out[i] = (bf16_t)(in[i] * scale);
}

// ---------------------------------------------------------------------------
// K4: QKV projection GEMM.  C(16384x3072) = x(16384x1024 f32) @ WqkvT^T.
// WqkvT is 3072x1024 bf16 (N-major).  Wave = 64x64 tile, block = 128x256.
// Epilogue scatters bf16 into q[b,h,n,d], k[b,h,n,d], vT[b,h,d,n].
// ---------------------------------------------------------------------------
__global__ __launch_bounds__(256, 1)
void k_gemm_qkv(const float* __restrict__ x, const bf16_t* __restrict__ wT,
                bf16_t* __restrict__ qb, bf16_t* __restrict__ kb,
                bf16_t* __restrict__ vT) {
  const int lane = threadIdx.x & 31;
  const int w    = threadIdx.x >> 5;
  const int l15  = lane & 15;
  const int rb   = (lane >> 4) * 8;    // C-layout row base
  const int kb2  = (lane >> 4) * 16;   // B kb
  const int M0 = blockIdx.y * 128 + (w >> 2) * 64;
  const int N0 = blockIdx.x * 256 + (w & 3) * 64;

  v8f acc[4][4] = {};
  for (int k0 = 0; k0 < DIM; k0 += 32) {
    if (k0 + 32 < DIM) {
      __builtin_prefetch(x  + (size_t)(M0 + l15) * DIM + k0 + 32, 0, 0);
      __builtin_prefetch(wT + (size_t)(N0 + l15) * DIM + k0 + 32, 0, 0);
    }
    v16bf af[4], bfv[4];
#pragma unroll
    for (int i = 0; i < 4; ++i)
      af[i] = ldA_f32(x + (size_t)(M0 + i * 16 + l15) * DIM + k0 + kb2);
#pragma unroll
    for (int j = 0; j < 4; ++j)
      bfv[j] = *(const v16bf*)(wT + (size_t)(N0 + j * 16 + l15) * DIM + k0 + kb2);
#pragma unroll
    for (int i = 0; i < 4; ++i)
#pragma unroll
      for (int j = 0; j < 4; ++j)
        acc[i][j] = WMMA_BF16(af[i], bfv[j], acc[i][j]);
  }
  // scatter epilogue: columns [0,1024)=q, [1024,2048)=k, [2048,3072)=v (as vT)
#pragma unroll
  for (int i = 0; i < 4; ++i)
#pragma unroll
    for (int j = 0; j < 4; ++j)
#pragma unroll
      for (int r = 0; r < 8; ++r) {
        const int gm = M0 + i * 16 + rb + r;
        const int gn = N0 + j * 16 + l15;
        const int bb = gm >> 12, nrow = gm & (NTOK - 1);
        const int t = gn >> 10, rem = gn & (DIM - 1);
        const int head = rem >> 6, dd = rem & (DHEAD - 1);
        const size_t bh = (size_t)bb * NHEAD + head;
        const bf16_t v = (bf16_t)acc[i][j][r];
        if (t == 0)      qb[(bh * NTOK + nrow) * DHEAD + dd] = v;
        else if (t == 1) kb[(bh * NTOK + nrow) * DHEAD + dd] = v;
        else             vT[(bh * DHEAD + dd) * NTOK + nrow] = v;
      }
}

// ---------------------------------------------------------------------------
// K5: ak path, flash style.  Per (b,h): S = a(128x64) @ k^T (softmax over
// n=4096 streamed in chunks of 32), out1 = attn @ v.  One block per (b,h),
// 8 waves, wave = 16 m-rows x all 64 d-cols.  Writes out1^T[b,h,d,m] bf16.
// NOTE: the P-staging LDS slab is wave-private -> no __syncthreads needed;
// intra-wave LDS ordering is guaranteed (DS ops in-order + s_wait_dscnt).
// ---------------------------------------------------------------------------
__global__ __launch_bounds__(256, 1)
void k_attn_ak(const bf16_t* __restrict__ ab, const bf16_t* __restrict__ kb,
               const bf16_t* __restrict__ vT, bf16_t* __restrict__ o1T) {
  __shared__ bf16_t plds[8][16][32];   // per-wave P staging (C-layout -> A-layout)
  const int lane = threadIdx.x & 31;
  const int w    = threadIdx.x >> 5;
  const int l15  = lane & 15;
  const int rb   = (lane >> 4) * 8;
  const int kb2  = (lane >> 4) * 16;
  const int bh = blockIdx.x;           // 0..63
  const int h  = bh & (NHEAD - 1);
  const bf16_t* Ap = ab  + (size_t)h  * MAGENT * DHEAD;
  const bf16_t* Kp = kb  + (size_t)bh * NTOK * DHEAD;
  const bf16_t* Vp = vT  + (size_t)bh * DHEAD * NTOK;
  bf16_t*       Op = o1T + (size_t)bh * DHEAD * MAGENT;
  const int m0 = w * 16;

  v16bf afr[2];
#pragma unroll
  for (int kk = 0; kk < 2; ++kk)
    afr[kk] = ldA_bf16(Ap + (size_t)(m0 + l15) * DHEAD + kk * 32 + kb2);

  float rm[8], rl[8];
#pragma unroll
  for (int r = 0; r < 8; ++r) { rm[r] = -1e30f; rl[r] = 0.0f; }
  v8f oacc[4] = {};

  for (int nc = 0; nc < NTOK; nc += 32) {
    v8f s[2] = {};
#pragma unroll
    for (int t = 0; t < 2; ++t) {
      const int col = nc + t * 16 + l15;   // n index (B column)
#pragma unroll
      for (int kk = 0; kk < 2; ++kk) {
        v16bf bfr = *(const v16bf*)(Kp + (size_t)col * DHEAD + kk * 32 + kb2);
        s[t] = WMMA_BF16(afr[kk], bfr, s[t]);
      }
    }
    // online softmax update, rows rb..rb+7 of this wave's m-strip
#pragma unroll
    for (int r = 0; r < 8; ++r) {
      const float cm = hmax16(fmaxf(s[0][r], s[1][r]));
      const float nm = fmaxf(rm[r], cm);
      const float alpha = __expf(rm[r] - nm);
      rm[r] = nm;
      const float p0 = __expf(s[0][r] - nm);
      const float p1 = __expf(s[1][r] - nm);
      s[0][r] = p0; s[1][r] = p1;
      rl[r] = rl[r] * alpha + hsum16(p0 + p1);
#pragma unroll
      for (int t2 = 0; t2 < 4; ++t2) oacc[t2][r] *= alpha;
    }
    // C-layout P -> wave-private LDS slab (row-major 16x32 bf16)
#pragma unroll
    for (int t = 0; t < 2; ++t)
#pragma unroll
      for (int r = 0; r < 8; ++r)
        plds[w][rb + r][t * 16 + l15] = (bf16_t)s[t][r];
    v16bf pf = ldA_bf16(&plds[w][l15][kb2]);   // A-fragment, K = 32
#pragma unroll
    for (int t2 = 0; t2 < 4; ++t2) {
      const int dj = t2 * 16 + l15;
      v16bf bfr = *(const v16bf*)(Vp + (size_t)dj * NTOK + nc + kb2);
      oacc[t2] = WMMA_BF16(pf, bfr, oacc[t2]);
    }
  }
#pragma unroll
  for (int t2 = 0; t2 < 4; ++t2) {
    const int dj = t2 * 16 + l15;
#pragma unroll
    for (int r = 0; r < 8; ++r)
      Op[(size_t)dj * MAGENT + m0 + rb + r] = (bf16_t)(oacc[t2][r] / rl[r]);
  }
}

// ---------------------------------------------------------------------------
// K6: qa path.  Per wave: 16 q-rows; S = q @ a^T (16x128), full softmax over
// m=128, out2 = P @ out1 (via out1^T).  Writes out2 bf16 [b,n,h,d] row-major
// (A-operand layout for the final GEMM).  2048 blocks x 8 waves.
// ---------------------------------------------------------------------------
__global__ __launch_bounds__(256, 1)
void k_attn_qa(const bf16_t* __restrict__ qb, const bf16_t* __restrict__ ab,
               const bf16_t* __restrict__ o1T, bf16_t* __restrict__ out2) {
  __shared__ bf16_t plds[8][16][128];  // 32 KB, wave-private slabs
  const int lane = threadIdx.x & 31;
  const int w    = threadIdx.x >> 5;
  const int l15  = lane & 15;
  const int rb   = (lane >> 4) * 8;
  const int kb2  = (lane >> 4) * 16;
  const int bh = blockIdx.x >> 5;              // 32 blocks per (b,h)
  const int sb = blockIdx.x & 31;
  const int h  = bh & (NHEAD - 1), bb = bh >> 4;
  const int n0 = (sb * 8 + w) * 16;
  const bf16_t* Qp = qb  + (size_t)bh * NTOK * DHEAD;
  const bf16_t* Aa = ab  + (size_t)h  * MAGENT * DHEAD;
  const bf16_t* O1 = o1T + (size_t)bh * DHEAD * MAGENT;

  v16bf qf[2];
#pragma unroll
  for (int kk = 0; kk < 2; ++kk)
    qf[kk] = ldA_bf16(Qp + (size_t)(n0 + l15) * DHEAD + kk * 32 + kb2);

  v8f s[8] = {};
#pragma unroll
  for (int t = 0; t < 8; ++t) {
    const int col = t * 16 + l15;              // m index
#pragma unroll
    for (int kk = 0; kk < 2; ++kk) {
      v16bf bfr = *(const v16bf*)(Aa + (size_t)col * DHEAD + kk * 32 + kb2);
      s[t] = WMMA_BF16(qf[kk], bfr, s[t]);
    }
  }
  float rs[8];
#pragma unroll
  for (int r = 0; r < 8; ++r) {
    float mx = s[0][r];
#pragma unroll
    for (int t = 1; t < 8; ++t) mx = fmaxf(mx, s[t][r]);
    mx = hmax16(mx);
    float sum = 0.0f;
#pragma unroll
    for (int t = 0; t < 8; ++t) { const float p = __expf(s[t][r] - mx); s[t][r] = p; sum += p; }
    rs[r] = hsum16(sum);
  }
#pragma unroll
  for (int t = 0; t < 8; ++t)
#pragma unroll
    for (int r = 0; r < 8; ++r)
      plds[w][rb + r][t * 16 + l15] = (bf16_t)s[t][r];
  // wave-private slab: intra-wave DS ordering suffices, no barrier needed
  v16bf pf[4];
#pragma unroll
  for (int kk = 0; kk < 4; ++kk)
    pf[kk] = ldA_bf16(&plds[w][l15][kk * 32 + kb2]);

  v8f acc[4] = {};
#pragma unroll
  for (int t2 = 0; t2 < 4; ++t2) {
    const int dj = t2 * 16 + l15;
#pragma unroll
    for (int kk = 0; kk < 4; ++kk) {
      v16bf bfr = *(const v16bf*)(O1 + (size_t)dj * MAGENT + kk * 32 + kb2);
      acc[t2] = WMMA_BF16(pf[kk], bfr, acc[t2]);
    }
  }
#pragma unroll
  for (int t2 = 0; t2 < 4; ++t2) {
    const int dj = t2 * 16 + l15;
#pragma unroll
    for (int r = 0; r < 8; ++r) {
      const int nrow = n0 + rb + r;
      out2[((size_t)bb * NTOK + nrow) * DIM + h * DHEAD + dj] = (bf16_t)(acc[t2][r] / rs[r]);
    }
  }
}

// ---------------------------------------------------------------------------
// K7: output projection.  out(16384x1024 f32) = out2(bf16) @ WoutT^T.
// ---------------------------------------------------------------------------
__global__ __launch_bounds__(256, 1)
void k_gemm_out(const bf16_t* __restrict__ A, const bf16_t* __restrict__ wT,
                float* __restrict__ out) {
  const int lane = threadIdx.x & 31;
  const int w    = threadIdx.x >> 5;
  const int l15  = lane & 15;
  const int rb   = (lane >> 4) * 8;
  const int kb2  = (lane >> 4) * 16;
  const int M0 = blockIdx.y * 128 + (w >> 2) * 64;
  const int N0 = blockIdx.x * 256 + (w & 3) * 64;

  v8f acc[4][4] = {};
  for (int k0 = 0; k0 < DIM; k0 += 32) {
    if (k0 + 32 < DIM) {
      __builtin_prefetch(A  + (size_t)(M0 + l15) * DIM + k0 + 32, 0, 0);
      __builtin_prefetch(wT + (size_t)(N0 + l15) * DIM + k0 + 32, 0, 0);
    }
    v16bf af[4], bfv[4];
#pragma unroll
    for (int i = 0; i < 4; ++i)
      af[i] = ldA_bf16(A + (size_t)(M0 + i * 16 + l15) * DIM + k0 + kb2);
#pragma unroll
    for (int j = 0; j < 4; ++j)
      bfv[j] = *(const v16bf*)(wT + (size_t)(N0 + j * 16 + l15) * DIM + k0 + kb2);
#pragma unroll
    for (int i = 0; i < 4; ++i)
#pragma unroll
      for (int j = 0; j < 4; ++j)
        acc[i][j] = WMMA_BF16(af[i], bfv[j], acc[i][j]);
  }
#pragma unroll
  for (int i = 0; i < 4; ++i)
#pragma unroll
    for (int j = 0; j < 4; ++j)
#pragma unroll
      for (int r = 0; r < 8; ++r)
        out[(size_t)(M0 + i * 16 + rb + r) * DIM + N0 + j * 16 + l15] = acc[i][j][r];
}

// ---------------------------------------------------------------------------
// launch
// ---------------------------------------------------------------------------
extern "C" void kernel_launch(void* const* d_in, const int* in_sizes, int n_in,
                              void* d_out, int out_size, void* d_ws, size_t ws_size,
                              hipStream_t stream) {
  (void)in_sizes; (void)n_in; (void)out_size; (void)ws_size;
  const float* x     = (const float*)d_in[0];  // (4,4096,1024)
  const float* Wqkv  = (const float*)d_in[1];  // (1024,3072)
  const float* agent = (const float*)d_in[2];  // (16,128,64)
  const float* Wout  = (const float*)d_in[3];  // (1024,1024)
  float* out = (float*)d_out;

  char* ws = (char*)d_ws;
  bf16_t* wqkvT = (bf16_t*)(ws + 0);           //  6,291,456 B : Wqkv^T  (3072x1024)
  bf16_t* woutT = (bf16_t*)(ws + 6291456);     //  2,097,152 B : Wout^T  (1024x1024)
  bf16_t* abuf  = (bf16_t*)(ws + 8388608);     //    262,144 B : a*scale (16,128,64)
  bf16_t* qb    = (bf16_t*)(ws + 8650752);     // 33,554,432 B : q  (b,h,n,d)
  bf16_t* kbuf  = (bf16_t*)(ws + 42205184);    // 33,554,432 B : k  (b,h,n,d)
  bf16_t* vT    = (bf16_t*)(ws + 75759616);    // 33,554,432 B : v^T(b,h,d,n)
  bf16_t* o1T   = (bf16_t*)(ws + 109314048);   //  1,048,576 B : out1^T (b,h,d,m)
  bf16_t* out2b = (bf16_t*)(ws + 110362624);   // 33,554,432 B : out2 (b,n,h*d)

  k_transpose_bf16<<<dim3(NQKV / 32, DIM / 32), dim3(32, 8), 0, stream>>>(Wqkv, wqkvT, DIM, NQKV);
  k_transpose_bf16<<<dim3(DIM / 32, DIM / 32), dim3(32, 8), 0, stream>>>(Wout, woutT, DIM, DIM);
  k_scale_bf16<<<(NHEAD * MAGENT * DHEAD + 255) / 256, 256, 0, stream>>>(
      agent, abuf, NHEAD * MAGENT * DHEAD, 0.125f);  // SCALE = 64^-0.5
  k_gemm_qkv<<<dim3(NQKV / 256, MROWS / 128), 256, 0, stream>>>(x, wqkvT, qb, kbuf, vT);
  k_attn_ak<<<NBATCH * NHEAD, 256, 0, stream>>>(abuf, kbuf, vT, o1T);
  k_attn_qa<<<NBATCH * NHEAD * (NTOK / 16) / 8, 256, 0, stream>>>(qb, abuf, o1T, out2b);
  k_gemm_out<<<dim3(DIM / 256, MROWS / 128), 256, 0, stream>>>(out2b, woutT, out);
}
